// DANetModule_56891136803467
// MI455X (gfx1250) — matrix operand order repaired and checked by reading, not compiled, not run on hardware
//
#include <hip/hip_runtime.h>
#include <hip/hip_bf16.h>

// DANet dual attention (PAM flash-attention + CAM) for B=4,C=512,H=W=64 on gfx1250.
// All heavy GEMMs via v_wmma_f32_16x16x32_bf16 (wave32). Workspace ~126MB < 192MB L2.

#define DEVINL __device__ __forceinline__

typedef __bf16 bf16_t;
typedef __attribute__((ext_vector_type(16))) __bf16 v16bf;
typedef __attribute__((ext_vector_type(8)))  __bf16 v8bf;
typedef __attribute__((ext_vector_type(8)))  float  v8f;
typedef __attribute__((ext_vector_type(4)))  float  v4f;

constexpr int Bsz  = 4;
constexpr int Cch  = 512;
constexpr int Npix = 64 * 64;   // 4096
constexpr int Dq   = Cch / 8;   // 64

DEVINL v8f wmma_bf16(v16bf a, v16bf b, v8f c) {
  // 8 args: (neg_a, A, neg_b, B, c_mod, C, reuse_a, reuse_b)
  return __builtin_amdgcn_wmma_f32_16x16x32_bf16(false, a, false, b, (short)0, c,
                                                 false, false);
}

DEVINL v16bf cat8(v8bf lo, v8bf hi) {
  return __builtin_shufflevector(lo, hi, 0, 1, 2, 3, 4, 5, 6, 7, 8, 9, 10, 11, 12, 13, 14, 15);
}

// A fragment (16x32 bf16, row-major, lda elems): lane<16 -> row=lane, K {0..7,16..23};
// lane>=16 -> row=lane-16, K {8..15,24..31}.  Two 16B loads.
DEVINL v16bf load_a_bf16(const bf16_t* A, int lda, int row0, int k0, int lane) {
  const int r  = row0 + (lane & 15);
  const int kk = k0 + ((lane >> 4) << 3);
  const bf16_t* p = A + (size_t)r * lda + kk;
  return cat8(*(const v8bf*)p, *(const v8bf*)(p + 16));
}

// Same A fragment but from f32 weights, converted to bf16 on the fly.
DEVINL v16bf load_a_f32(const float* A, int lda, int row0, int k0, int lane) {
  const int r  = row0 + (lane & 15);
  const int kk = k0 + ((lane >> 4) << 3);
  const float* p = A + (size_t)r * lda + kk;
  v4f a0 = *(const v4f*)(p);      v4f a1 = *(const v4f*)(p + 4);
  v4f a2 = *(const v4f*)(p + 16); v4f a3 = *(const v4f*)(p + 20);
  v8f lo = __builtin_shufflevector(a0, a1, 0, 1, 2, 3, 4, 5, 6, 7);
  v8f hi = __builtin_shufflevector(a2, a3, 0, 1, 2, 3, 4, 5, 6, 7);
  return cat8(__builtin_convertvector(lo, v8bf), __builtin_convertvector(hi, v8bf));
}

// B fragment (32x16 bf16) from column-contiguous storage Bt[col][k] (ldb elems per col):
// lane<16 -> col=col0+lane, K {0..7,16..23}; lane>=16 -> col=col0+lane-16, K {8..15,24..31}.
DEVINL v16bf load_b_bf16(const bf16_t* Bt, int ldb, int col0, int k0, int lane) {
  const int c  = col0 + (lane & 15);
  const int kk = k0 + ((lane >> 4) << 3);
  const bf16_t* p = Bt + (size_t)c * ldb + kk;
  return cat8(*(const v8bf*)p, *(const v8bf*)(p + 16));
}

// ---------------- x [B,C,N] f32  ->  xbT [B,N,C] bf16 (LDS tile transpose) -----------
__global__ void k_cvt_transpose(const float* __restrict__ x, bf16_t* __restrict__ xbT) {
  __shared__ float tile[32][33];
  const int b = blockIdx.z;
  const int c0 = blockIdx.y << 5, n0 = blockIdx.x << 5;
  const int tx = threadIdx.x, ty = threadIdx.y;
#pragma unroll
  for (int i = ty; i < 32; i += 8)
    tile[i][tx] = x[(size_t)b * Cch * Npix + (size_t)(c0 + i) * Npix + n0 + tx];
  __syncthreads();
#pragma unroll
  for (int i = ty; i < 32; i += 8)
    xbT[(size_t)b * Npix * Cch + (size_t)(n0 + i) * Cch + c0 + tx] = (bf16_t)tile[tx][i];
}

// ---------------- 1x1-conv projection: out = W[M,C] * x[C,N] + bias --------------------
// TOUT=true  -> out stored [B,N,M] bf16 (n-major, contiguous 16B store per lane)
// TOUT=false -> out stored [B,M,N] bf16
template <int MTOT, bool TOUT>
__global__ void __launch_bounds__(256) k_proj(const float* __restrict__ W,
                                              const float* __restrict__ bias,
                                              const bf16_t* __restrict__ xbT,
                                              bf16_t* __restrict__ out) {
  const int lane = threadIdx.x & 31;
  const int wave = threadIdx.x >> 5;                 // 8 waves: 2(m) x 4(n) tiles
  const int b  = blockIdx.z;
  const int n0 = blockIdx.x * 64 + (wave >> 1) * 16;
  const int m0 = blockIdx.y * 32 + (wave & 1) * 16;
  const bf16_t* X = xbT + (size_t)b * Npix * Cch;
  v8f acc = {};
#pragma unroll 4
  for (int k0 = 0; k0 < Cch; k0 += 32)
    acc = wmma_bf16(load_a_f32(W, Cch, m0, k0, lane),
                    load_b_bf16(X, Cch, n0, k0, lane), acc);
  const int rb = m0 + ((lane >> 4) << 3);            // C-layout: VGPR v -> row rb+v
  const int cc = n0 + (lane & 15);                   // lane -> column
  if (TOUT) {
    v8bf ov;
#pragma unroll
    for (int v = 0; v < 8; ++v) ov[v] = (bf16_t)(acc[v] + bias[rb + v]);
    *(v8bf*)(out + (size_t)b * Npix * MTOT + (size_t)cc * MTOT + rb) = ov;
  } else {
#pragma unroll
    for (int v = 0; v < 8; ++v)
      out[(size_t)b * MTOT * Npix + (size_t)(rb + v) * Npix + cc] =
          (bf16_t)(acc[v] + bias[rb + v]);
  }
}

// ---------------- generic bf16 GEMM, f32 out (used for CAM energy / CAM out) ----------
__global__ void __launch_bounds__(256) k_gemm_bf16(const bf16_t* __restrict__ A, size_t strA, int lda,
                                                   const bf16_t* __restrict__ Bt, size_t strB, int ldb,
                                                   float* __restrict__ Cm, size_t strC, int ldc,
                                                   int Kd) {
  const int lane = threadIdx.x & 31;
  const int wave = threadIdx.x >> 5;
  const int b  = blockIdx.z;
  const int n0 = blockIdx.x * 64 + (wave >> 1) * 16;
  const int m0 = blockIdx.y * 32 + (wave & 1) * 16;
  const bf16_t* Ab = A + (size_t)b * strA;
  const bf16_t* Bb = Bt + (size_t)b * strB;
  v8f acc = {};
#pragma unroll 4
  for (int k0 = 0; k0 < Kd; k0 += 32)
    acc = wmma_bf16(load_a_bf16(Ab, lda, m0, k0, lane),
                    load_b_bf16(Bb, ldb, n0, k0, lane), acc);
  const int rb = m0 + ((lane >> 4) << 3);
  const int cc = n0 + (lane & 15);
  float* Cb = Cm + (size_t)b * strC;
#pragma unroll
  for (int v = 0; v < 8; ++v) Cb[(size_t)(rb + v) * ldc + cc] = acc[v];
}

// ---------------- PAM flash attention --------------------------------------------------
// qpT,kpT: [B,N,64] bf16 ; vp: [B,C,N] bf16 ; pamO: [B,C,N] f32 (pre-gamma)
// Wave owns 16 q-rows x 512 channels of O (32 v8f accumulators); online softmax over k.
__global__ void __launch_bounds__(256, 1) k_pam_flash(const bf16_t* __restrict__ qpT,
                                                      const bf16_t* __restrict__ kpT,
                                                      const bf16_t* __restrict__ vp,
                                                      float* __restrict__ pamO) {
  __shared__ bf16_t pbuf[8][16 * 32];                // wave-private P staging (C->A relayout)
  const int lane = threadIdx.x & 31;
  const int wave = threadIdx.x >> 5;
  const int b  = blockIdx.y;
  const int q0 = blockIdx.x * 128 + wave * 16;
  const bf16_t* Q  = qpT + (size_t)b * Npix * Dq;
  const bf16_t* Kp = kpT + (size_t)b * Npix * Dq;
  const bf16_t* V  = vp  + (size_t)b * Cch * Npix;

  const v16bf qa0 = load_a_bf16(Q, Dq, q0, 0, lane);   // d = 0..31
  const v16bf qa1 = load_a_bf16(Q, Dq, q0, 32, lane);  // d = 32..63

  v8f o[32];
#pragma unroll
  for (int t = 0; t < 32; ++t) o[t] = {};
  float mrun[8], lrun[8];
#pragma unroll
  for (int v = 0; v < 8; ++v) { mrun[v] = -3.0e38f; lrun[v] = 0.f; }

  bf16_t* pb = &pbuf[wave][0];

  for (int k0 = 0; k0 < Npix; k0 += 32) {
    // S tile: 16 q x 32 k  (two 16x16 tiles, K-dim = 64 channels)
    v8f s0 = {}, s1 = {};
    s0 = wmma_bf16(qa0, load_b_bf16(Kp, Dq, k0,       0, lane), s0);
    s0 = wmma_bf16(qa1, load_b_bf16(Kp, Dq, k0,      32, lane), s0);
    s1 = wmma_bf16(qa0, load_b_bf16(Kp, Dq, k0 + 16,  0, lane), s1);
    s1 = wmma_bf16(qa1, load_b_bf16(Kp, Dq, k0 + 16, 32, lane), s1);

    // online softmax; rows of each lane-half are disjoint -> reduce within 16-lane halves
    float p0[8], p1[8];
#pragma unroll
    for (int v = 0; v < 8; ++v) {
      float mv = fmaxf(s0[v], s1[v]);
      mv = fmaxf(mv, __shfl_xor(mv, 1, 32));
      mv = fmaxf(mv, __shfl_xor(mv, 2, 32));
      mv = fmaxf(mv, __shfl_xor(mv, 4, 32));
      mv = fmaxf(mv, __shfl_xor(mv, 8, 32));
      const float mnew = fmaxf(mrun[v], mv);
      const float scl  = __expf(mrun[v] - mnew);
      p0[v] = __expf(s0[v] - mnew);
      p1[v] = __expf(s1[v] - mnew);
      float rs = p0[v] + p1[v];
      rs += __shfl_xor(rs, 1, 32);
      rs += __shfl_xor(rs, 2, 32);
      rs += __shfl_xor(rs, 4, 32);
      rs += __shfl_xor(rs, 8, 32);
      lrun[v] = lrun[v] * scl + rs;
      mrun[v] = mnew;
#pragma unroll
      for (int t = 0; t < 32; ++t) o[t][v] *= scl;   // rescale O row v
    }

    // P (C-layout) -> LDS -> reload as A fragment (DS ops are in-order per wave)
    const int prow = (lane >> 4) << 3;
    const int pcol = lane & 15;
#pragma unroll
    for (int v = 0; v < 8; ++v) {
      pb[(prow + v) * 32 + pcol]      = (bf16_t)p0[v];
      pb[(prow + v) * 32 + 16 + pcol] = (bf16_t)p1[v];
    }
    const v16bf pa = load_a_bf16(pb, 32, 0, 0, lane);

    // O += P (16x32) x V^T (32x16 per c-tile); V[c][k] is column-contiguous B operand
#pragma unroll
    for (int t = 0; t < 32; ++t)
      o[t] = wmma_bf16(pa, load_b_bf16(V, Npix, t * 16, k0, lane), o[t]);
  }

  const int rb  = (lane >> 4) << 3;
  const int cc0 = lane & 15;
#pragma unroll
  for (int v = 0; v < 8; ++v) {
    const float inv = 1.f / lrun[v];
    const int q = q0 + rb + v;
#pragma unroll
    for (int t = 0; t < 32; ++t)
      pamO[(size_t)b * Cch * Npix + (size_t)(t * 16 + cc0) * Npix + q] = o[t][v] * inv;
  }
}

// ---------------- CAM softmax of (rowmax - e): attn = exp(rowmin - e)/sum --------------
__global__ void __launch_bounds__(256) k_cam_softmax(const float* __restrict__ energy,
                                                     bf16_t* __restrict__ attn) {
  const int lane = threadIdx.x & 31;
  const int wave = threadIdx.x >> 5;
  const int row = blockIdx.x * 8 + wave;
  const int b = blockIdx.y;
  const float* e = energy + ((size_t)b * Cch + row) * Cch;
  bf16_t* a = attn + ((size_t)b * Cch + row) * Cch;
  float ev[16];
  float mn = 3.0e38f;
#pragma unroll
  for (int j = 0; j < 16; ++j) { ev[j] = e[lane + j * 32]; mn = fminf(mn, ev[j]); }
#pragma unroll
  for (int s = 1; s < 32; s <<= 1) mn = fminf(mn, __shfl_xor(mn, s, 32));
  float sum = 0.f;
#pragma unroll
  for (int j = 0; j < 16; ++j) { ev[j] = __expf(mn - ev[j]); sum += ev[j]; }
#pragma unroll
  for (int s = 1; s < 32; s <<= 1) sum += __shfl_xor(sum, s, 32);
  const float inv = 1.f / sum;
#pragma unroll
  for (int j = 0; j < 16; ++j) a[lane + j * 32] = (bf16_t)(ev[j] * inv);
}

// ---------------- final combine: out = gp*pamO + gc*camO + 2*x -------------------------
__global__ void k_combine(const float* __restrict__ x, const float* __restrict__ pamO,
                          const float* __restrict__ camO, const float* __restrict__ pg,
                          const float* __restrict__ cg, float* __restrict__ out) {
  const size_t i = (size_t)blockIdx.x * blockDim.x + threadIdx.x;
  out[i] = pg[0] * pamO[i] + cg[0] * camO[i] + 2.f * x[i];
}

extern "C" void kernel_launch(void* const* d_in, const int* in_sizes, int n_in,
                              void* d_out, int out_size, void* d_ws, size_t ws_size,
                              hipStream_t stream) {
  const float* x      = (const float*)d_in[0];
  const float* pam_wq = (const float*)d_in[1];
  const float* pam_bq = (const float*)d_in[2];
  const float* pam_wk = (const float*)d_in[3];
  const float* pam_bk = (const float*)d_in[4];
  const float* pam_wv = (const float*)d_in[5];
  const float* pam_bv = (const float*)d_in[6];
  const float* pam_g  = (const float*)d_in[7];
  const float* cam_wq = (const float*)d_in[8];
  const float* cam_bq = (const float*)d_in[9];
  const float* cam_wk = (const float*)d_in[10];
  const float* cam_bk = (const float*)d_in[11];
  const float* cam_wv = (const float*)d_in[12];
  const float* cam_bv = (const float*)d_in[13];
  const float* cam_g  = (const float*)d_in[14];
  float* out = (float*)d_out;

  char* ws = (char*)d_ws;
  size_t off = 0;
  auto wsa = [&](size_t bytes) -> char* {
    char* p = ws + off;
    off += (bytes + 255) & ~(size_t)255;
    return p;
  };
  bf16_t* xbT = (bf16_t*)wsa((size_t)Bsz * Npix * Cch * 2);  // x^T bf16      16 MB
  bf16_t* qpT = (bf16_t*)wsa((size_t)Bsz * Npix * Dq  * 2);  // PAM Q^T        2 MB
  bf16_t* kpT = (bf16_t*)wsa((size_t)Bsz * Npix * Dq  * 2);  // PAM K^T        2 MB
  bf16_t* vp  = (bf16_t*)wsa((size_t)Bsz * Cch * Npix * 2);  // PAM V         16 MB
  bf16_t* qc  = (bf16_t*)wsa((size_t)Bsz * Dq  * Npix * 2);  // CAM Q (flat [512,512])
  bf16_t* kc  = (bf16_t*)wsa((size_t)Bsz * Dq  * Npix * 2);  // CAM K
  bf16_t* vcT = (bf16_t*)wsa((size_t)Bsz * Npix * Cch * 2);  // CAM V^T       16 MB
  float*  eC  = (float*) wsa((size_t)Bsz * Cch * Cch * 4);   // CAM energy     4 MB
  bf16_t* aC  = (bf16_t*)wsa((size_t)Bsz * Cch * Cch * 2);   // CAM attn       2 MB
  float*  pO  = (float*) wsa((size_t)Bsz * Cch * Npix * 4);  // PAM out       32 MB
  float*  cO  = (float*) wsa((size_t)Bsz * Cch * Npix * 4);  // CAM out       32 MB
  (void)ws_size; (void)in_sizes; (void)n_in; (void)out_size;

  // 0) transpose+convert x
  k_cvt_transpose<<<dim3(Npix / 32, Cch / 32, Bsz), dim3(32, 8), 0, stream>>>(x, xbT);

  // 1) six 1x1-conv projections (WMMA GEMM, f32 weights converted in fragment load)
  k_proj<Dq,  true ><<<dim3(Npix / 64, Dq  / 32, Bsz), 256, 0, stream>>>(pam_wq, pam_bq, xbT, qpT);
  k_proj<Dq,  true ><<<dim3(Npix / 64, Dq  / 32, Bsz), 256, 0, stream>>>(pam_wk, pam_bk, xbT, kpT);
  k_proj<Cch, false><<<dim3(Npix / 64, Cch / 32, Bsz), 256, 0, stream>>>(pam_wv, pam_bv, xbT, vp);
  k_proj<Dq,  false><<<dim3(Npix / 64, Dq  / 32, Bsz), 256, 0, stream>>>(cam_wq, cam_bq, xbT, qc);
  k_proj<Dq,  false><<<dim3(Npix / 64, Dq  / 32, Bsz), 256, 0, stream>>>(cam_wk, cam_bk, xbT, kc);
  k_proj<Cch, true ><<<dim3(Npix / 64, Cch / 32, Bsz), 256, 0, stream>>>(cam_wv, cam_bv, xbT, vcT);

  // 2) PAM flash attention (no NxN materialization)
  k_pam_flash<<<dim3(Npix / 128, Bsz), 256, 0, stream>>>(qpT, kpT, vp, pO);

  // 3) CAM energy = q~ [512x512] x k~^T  (both flat views of [64,4096])
  k_gemm_bf16<<<dim3(Cch / 64, Cch / 32, Bsz), 256, 0, stream>>>(
      qc, (size_t)Cch * Cch, Cch, kc, (size_t)Cch * Cch, Cch,
      eC, (size_t)Cch * Cch, Cch, Cch);

  // 4) CAM softmax of (rowmax - e)
  k_cam_softmax<<<dim3(Cch / 8, Bsz), 256, 0, stream>>>(eC, aC);

  // 5) CAM out = attn [512x512] x v [512x4096] (B operand = vcT, column-contiguous)
  k_gemm_bf16<<<dim3(Npix / 64, Cch / 32, Bsz), 256, 0, stream>>>(
      aC, (size_t)Cch * Cch, Cch, vcT, (size_t)Npix * Cch, Cch,
      cO, (size_t)Cch * Npix, Npix, Cch);

  // 6) combine
  k_combine<<<dim3((Bsz * Cch * Npix) / 256), 256, 0, stream>>>(x, pO, cO, pam_g, cam_g, out);
}
